// BatteryTransformerBlock_35184372089393
// MI455X (gfx1250) — matrix operand order repaired and checked by reading, not compiled, not run on hardware
//
#include <hip/hip_runtime.h>
#include <hip/hip_bf16.h>
#include <math.h>

// ---------------------------------------------------------------------------
// BatteryTransformerBlock for MI455X (gfx1250, wave32, WMMA).
// fp32 reference via split-bf16 (hi/lo): 3x v_wmma_f32_16x16x32_bf16 per K=32
// slab ~ fp32 accuracy with fp32 accumulation. Conversion fp32->bf16 happens
// ONCE at LDS staging (pair-packed); WMMA fragments are pure ds_load_b128.
// GEMM K-loop is double-buffered: global loads of slab k+1 overlap WMMA on
// slab k with a single barrier per step.
// temporal/sensor biases are per-row softmax constants -> provably cancel.
// ---------------------------------------------------------------------------

typedef unsigned int u32;
typedef __attribute__((ext_vector_type(2)))  u32    u32x2;
typedef __attribute__((ext_vector_type(4)))  u32    u32x4;
typedef __attribute__((ext_vector_type(2)))  __bf16 bf16x2;
typedef __attribute__((ext_vector_type(16))) __bf16 bf16x16;
typedef __attribute__((ext_vector_type(8)))  float  f32x8;

#define SEQ     2048
#define DMODEL  512
#define NHEADS  8
#define DK      64
#define ROWS    4096   // B*S
#define DFF     2048

union Frag { bf16x16 v; u32x4 q[2]; };

// --------------------------- helpers ---------------------------------------

__device__ __forceinline__ void wmma3(f32x8& acc,
                                      const bf16x16 ahi, const bf16x16 alo,
                                      const bf16x16 bhi, const bf16x16 blo) {
  acc = __builtin_amdgcn_wmma_f32_16x16x32_bf16(false, ahi, false, bhi, (short)0, acc, false, false);
  acc = __builtin_amdgcn_wmma_f32_16x16x32_bf16(false, ahi, false, blo, (short)0, acc, false, false);
  acc = __builtin_amdgcn_wmma_f32_16x16x32_bf16(false, alo, false, bhi, (short)0, acc, false, false);
}

// split a,b (consecutive elements of one K-pair) into packed bf16 hi and lo
__device__ __forceinline__ void cvt_hilo(float a, float b, u32& hi, u32& lo) {
  const __bf16 ha = (__bf16)a, hb = (__bf16)b;
  bf16x2 h; h[0] = ha; h[1] = hb;
  bf16x2 l; l[0] = (__bf16)(a - (float)ha); l[1] = (__bf16)(b - (float)hb);
  hi = __builtin_bit_cast(u32, h);
  lo = __builtin_bit_cast(u32, l);
}

// A fragment (16x32 MxK) from pair-packed LDS tile [m][kpair], pitch in u32.
// lanes 0-15: rows M, K {0..7,16..23}; lanes 16-31: K {8..15,24..31}.
__device__ __forceinline__ bf16x16 load_a_frag_p(const u32* base, int pitch, int lane) {
  const int l15 = lane & 15, sel = (lane >> 4) & 1;
  const u32* row = base + l15 * pitch;
  Frag f;
  f.q[0] = *(const u32x4*)(row + sel * 4);
  f.q[1] = *(const u32x4*)(row + 8 + sel * 4);
  return f.v;
}

// B fragment (32x16 KxN) from pair-packed LDS tile [n][kpair], pitch in u32.
// lanes 0-15: col N, K {0..7,16..23}; lanes 16-31: K {8..15,24..31}.
__device__ __forceinline__ bf16x16 load_b_frag_p(const u32* base, int pitch, int lane) {
  const int l15 = lane & 15, sel = (lane >> 4) & 1;
  const u32* row = base + l15 * pitch;
  Frag f;
  f.q[0] = *(const u32x4*)(row + sel * 4);
  f.q[1] = *(const u32x4*)(row + 8 + sel * 4);
  return f.v;
}

// one-shot A fragment straight from an fp32 LDS tile (used for Q only)
__device__ __forceinline__ void build_a_frag_f32(const float* __restrict__ src, int pitch,
                                                 int lane, bf16x16& hi_f, bf16x16& lo_f) {
  const int l15 = lane & 15;
  const int sel = (lane >> 4) & 1;
  const float* row = src + l15 * pitch;
#pragma unroll
  for (int i = 0; i < 8; ++i) {
    const int k = ((i < 4) ? 0 : 16) + sel * 8 + (i & 3) * 2;
    const float a0 = row[k], a1 = row[k + 1];
    const __bf16 h0 = (__bf16)a0, h1 = (__bf16)a1;
    hi_f[2 * i] = h0;  hi_f[2 * i + 1] = h1;
    lo_f[2 * i] = (__bf16)(a0 - (float)h0);
    lo_f[2 * i + 1] = (__bf16)(a1 - (float)h1);
  }
}

__device__ __forceinline__ float wave_sum(float v) {
#pragma unroll
  for (int off = 16; off > 0; off >>= 1) v += __shfl_xor(v, off, 32);
  return v;
}
__device__ __forceinline__ float wave_max(float v) {
#pragma unroll
  for (int off = 16; off > 0; off >>= 1) v = fmaxf(v, __shfl_xor(v, off, 32));
  return v;
}
__device__ __forceinline__ float gelu_exact(float x) {
  return 0.5f * x * (1.0f + erff(x * 0.70710678118654752f));
}

// --------------------------- tiled GEMM ------------------------------------
// out[M,N] = act( A[M,K] * W[K,N] + bias ) (+ res[M,N])
// 256 threads (8 waves); block tile 128x64, K-step 32; wave w owns rows
// [w*16, w*16+16) x 64 cols. Double-buffered pair-packed bf16 hi/lo tiles.

__device__ __forceinline__ void stage_tiles(const float* __restrict__ A,
                                            const float* __restrict__ W,
                                            int K, int N, int m0, int n0, int kk, int tid,
                                            u32 (*Ahi)[20], u32 (*Alo)[20],
                                            u32 (*Bhi)[20], u32 (*Blo)[20]) {
  // A tile 128x32 fp32 -> pair-packed bf16 hi/lo
#pragma unroll
  for (int s = tid; s < 1024; s += 256) {
    const int r  = s >> 3;
    const int c4 = s & 7;
    const float4 v = *(const float4*)(A + (size_t)(m0 + r) * K + kk + c4 * 4);
    u32 h0, l0, h1, l1;
    cvt_hilo(v.x, v.y, h0, l0);
    cvt_hilo(v.z, v.w, h1, l1);
    const u32x2 ph = {h0, h1}, pl = {l0, l1};
    *(u32x2*)(&Ahi[r][c4 * 2]) = ph;
    *(u32x2*)(&Alo[r][c4 * 2]) = pl;
  }
  // W tile 32x64 fp32 -> transposed [n][kpair] bf16 hi/lo
  {
    const int kp = tid >> 4;     // 0..15
    const int n4 = tid & 15;
    const float4 v0 = *(const float4*)(W + (size_t)(kk + 2 * kp)     * N + n0 + n4 * 4);
    const float4 v1 = *(const float4*)(W + (size_t)(kk + 2 * kp + 1) * N + n0 + n4 * 4);
    u32 h, l;
    cvt_hilo(v0.x, v1.x, h, l); Bhi[n4 * 4 + 0][kp] = h; Blo[n4 * 4 + 0][kp] = l;
    cvt_hilo(v0.y, v1.y, h, l); Bhi[n4 * 4 + 1][kp] = h; Blo[n4 * 4 + 1][kp] = l;
    cvt_hilo(v0.z, v1.z, h, l); Bhi[n4 * 4 + 2][kp] = h; Blo[n4 * 4 + 2][kp] = l;
    cvt_hilo(v0.w, v1.w, h, l); Bhi[n4 * 4 + 3][kp] = h; Blo[n4 * 4 + 3][kp] = l;
  }
}

template <int ACT, bool HASBIAS, bool RES>
__global__ __launch_bounds__(256)
void gemm_kernel(const float* __restrict__ A, const float* __restrict__ W,
                 const float* __restrict__ bias, const float* __restrict__ res,
                 float* __restrict__ out, int M, int N, int K) {
  __shared__ u32 Ahi[2][128][20];   // [buf][m][kpair], 16B-aligned rows
  __shared__ u32 Alo[2][128][20];
  __shared__ u32 Bhi[2][64][20];    // [buf][n][kpair]
  __shared__ u32 Blo[2][64][20];

  const int tid  = threadIdx.x;
  const int wave = tid >> 5;
  const int lane = tid & 31;
  const int m0   = blockIdx.y * 128;
  const int n0   = blockIdx.x * 64;

  const f32x8 zero = {0.f, 0.f, 0.f, 0.f, 0.f, 0.f, 0.f, 0.f};
  f32x8 acc[4] = {zero, zero, zero, zero};

  stage_tiles(A, W, K, N, m0, n0, 0, tid, Ahi[0], Alo[0], Bhi[0], Blo[0]);
  __syncthreads();

  const int nsteps = K >> 5;
  for (int step = 0; step < nsteps; ++step) {
    const int buf = step & 1;
    if (step + 1 < nsteps) {
      // prefetch slab k+2, stage slab k+1 into the other buffer while the
      // WMMA chain below consumes slab k (single barrier per step).
      if (step + 2 < nsteps) {
        const int kpf = (step + 2) * 32;
        if (tid < 128)      __builtin_prefetch(A + (size_t)(m0 + tid) * K + kpf, 0, 3);
        else if (tid < 160) __builtin_prefetch(W + (size_t)(kpf + (tid - 128)) * N + n0, 0, 3);
      }
      stage_tiles(A, W, K, N, m0, n0, (step + 1) * 32, tid,
                  Ahi[buf ^ 1], Alo[buf ^ 1], Bhi[buf ^ 1], Blo[buf ^ 1]);
    }

    const bf16x16 ahi = load_a_frag_p(&Ahi[buf][wave * 16][0], 20, lane);
    const bf16x16 alo = load_a_frag_p(&Alo[buf][wave * 16][0], 20, lane);
#pragma unroll
    for (int t = 0; t < 4; ++t) {
      const bf16x16 bhi = load_b_frag_p(&Bhi[buf][t * 16][0], 20, lane);
      const bf16x16 blo = load_b_frag_p(&Blo[buf][t * 16][0], 20, lane);
      wmma3(acc[t], ahi, alo, bhi, blo);
    }
    __syncthreads();
  }

  // epilogue: acc element r -> (m = sel*8+r, n = lane&15)
  const int l15 = lane & 15;
  const int sel = (lane >> 4) & 1;
#pragma unroll
  for (int t = 0; t < 4; ++t) {
    const int n  = n0 + t * 16 + l15;
    const float bv = HASBIAS ? bias[n] : 0.f;
#pragma unroll
    for (int r = 0; r < 8; ++r) {
      const int m = m0 + wave * 16 + sel * 8 + r;
      float v = acc[t][r] + bv;
      if (RES)      v += res[(size_t)m * N + n];
      if (ACT == 1) v = gelu_exact(v);
      out[(size_t)m * N + n] = v;
    }
  }
}

// --------------------------- attention -------------------------------------
// grid = (B*H, SEQ/16); 128 threads (4 waves). A workgroup owns 16 query rows
// of one (b,h): 16x2048 fp32 score tile resident in LDS, in-LDS masked
// softmax, attn written to HBM exactly once, then ctx = P * V via WMMA.

__global__ __launch_bounds__(128)
void attn_kernel(const float* __restrict__ Q, const float* __restrict__ Km,
                 const float* __restrict__ V, const int* __restrict__ mask,
                 float* __restrict__ attn, float* __restrict__ ctx) {
  __shared__ float P[16][SEQ + 4];     // scores -> probs (fp32)
  __shared__ float Qs[16][DK + 2];     // Q tile [q][d] fp32
  __shared__ u32   KVhi[64][36];       // scores: [k][dpair] ; ctx: [d][kpair]
  __shared__ u32   KVlo[64][36];
  __shared__ u32   Phs[16][36];        // P chunk, pair-packed bf16 hi
  __shared__ u32   Pls[16][36];        // ... lo

  const int tid  = threadIdx.x;
  const int wave = tid >> 5;
  const int lane = tid & 31;
  const int l15  = lane & 15;
  const int sel  = (lane >> 4) & 1;

  const int bh = blockIdx.x;           // 0..15
  const int b  = bh >> 3;
  const int h  = bh & 7;
  const int q0 = blockIdx.y * 16;

  // ---- stage Q tile (fp32, one time) ----
#pragma unroll
  for (int s = tid; s < 16 * 16; s += 128) {
    const int r  = s >> 4;
    const int d4 = s & 15;
    const float4 v = *(const float4*)(Q + (size_t)(b * SEQ + q0 + r) * DMODEL + h * DK + d4 * 4);
    Qs[r][d4 * 4 + 0] = v.x; Qs[r][d4 * 4 + 1] = v.y;
    Qs[r][d4 * 4 + 2] = v.z; Qs[r][d4 * 4 + 3] = v.w;
  }
  __syncthreads();

  bf16x16 qhi[2], qlo[2];
#pragma unroll
  for (int slab = 0; slab < 2; ++slab)
    build_a_frag_f32(&Qs[0][slab * 32], DK + 2, lane, qhi[slab], qlo[slab]);

  // ---- scores: S = Q K^T / sqrt(dk) ----
  for (int c = 0; c < SEQ / 64; ++c) {
    if (c + 1 < SEQ / 64 && tid < 64)
      __builtin_prefetch(Km + (size_t)(b * SEQ + (c + 1) * 64 + tid) * DMODEL + h * DK, 0, 3);
    // K chunk 64 x 64 fp32 -> [k][dpair] bf16 hi/lo
#pragma unroll
    for (int s = tid; s < 1024; s += 128) {
      const int kr = s >> 4;
      const int d4 = s & 15;
      const float4 v = *(const float4*)(Km + (size_t)(b * SEQ + c * 64 + kr) * DMODEL + h * DK + d4 * 4);
      u32 h0, l0, h1, l1;
      cvt_hilo(v.x, v.y, h0, l0);
      cvt_hilo(v.z, v.w, h1, l1);
      const u32x2 ph = {h0, h1}, pl = {l0, l1};
      *(u32x2*)(&KVhi[kr][d4 * 2]) = ph;
      *(u32x2*)(&KVlo[kr][d4 * 2]) = pl;
    }
    __syncthreads();

    f32x8 sc = {0.f, 0.f, 0.f, 0.f, 0.f, 0.f, 0.f, 0.f};
#pragma unroll
    for (int slab = 0; slab < 2; ++slab) {
      // B[d][n] with n = K-row inside this wave's 16-col tile
      const bf16x16 bhi = load_b_frag_p(&KVhi[wave * 16][0] + slab * 16, 36, lane);
      const bf16x16 blo = load_b_frag_p(&KVlo[wave * 16][0] + slab * 16, 36, lane);
      wmma3(sc, qhi[slab], qlo[slab], bhi, blo);
    }
    const int kcol = c * 64 + wave * 16 + l15;
#pragma unroll
    for (int r = 0; r < 8; ++r)
      P[sel * 8 + r][kcol] = sc[r] * 0.125f;   // 1/sqrt(64)
    __syncthreads();
  }

  // ---- masked softmax in LDS, single write of attn ----
#pragma unroll 1
  for (int rr = 0; rr < 4; ++rr) {
    const int q  = wave * 4 + rr;
    const int qg = q0 + q;
    const int* mrow = mask + (size_t)qg * SEQ;

    float mx = -3.0e38f;
    for (int i = lane; i < SEQ; i += 32) {
      float s = P[q][i];
      if (mrow[i] == 0) s = -1.0e9f;
      P[q][i] = s;
      mx = fmaxf(mx, s);
    }
    mx = wave_max(mx);

    float sum = 0.f;
    for (int i = lane; i < SEQ; i += 32) {
      const float p = __expf(P[q][i] - mx);
      P[q][i] = p;
      sum += p;
    }
    sum = wave_sum(sum);
    const float inv = 1.0f / sum;

    float* arow = attn + ((size_t)bh * SEQ + qg) * SEQ;
    for (int i = lane; i < SEQ; i += 32) {
      const float p = P[q][i] * inv;
      P[q][i] = p;
      arow[i] = p;
    }
  }
  __syncthreads();

  // ---- ctx = P * V : wave w makes the 16x16 tile at cols w*16.. ----
  f32x8 cacc = {0.f, 0.f, 0.f, 0.f, 0.f, 0.f, 0.f, 0.f};
  for (int c = 0; c < SEQ / 64; ++c) {
    if (c + 1 < SEQ / 64 && tid < 64)
      __builtin_prefetch(V + (size_t)(b * SEQ + (c + 1) * 64 + tid) * DMODEL + h * DK, 0, 3);
    // V chunk -> transposed [d][kpair] bf16 hi/lo
#pragma unroll
    for (int s = tid; s < 512; s += 128) {
      const int kp = s >> 4;
      const int d4 = s & 15;
      const float4 v0 = *(const float4*)(V + (size_t)(b * SEQ + c * 64 + 2 * kp)     * DMODEL + h * DK + d4 * 4);
      const float4 v1 = *(const float4*)(V + (size_t)(b * SEQ + c * 64 + 2 * kp + 1) * DMODEL + h * DK + d4 * 4);
      u32 hh, ll;
      cvt_hilo(v0.x, v1.x, hh, ll); KVhi[d4 * 4 + 0][kp] = hh; KVlo[d4 * 4 + 0][kp] = ll;
      cvt_hilo(v0.y, v1.y, hh, ll); KVhi[d4 * 4 + 1][kp] = hh; KVlo[d4 * 4 + 1][kp] = ll;
      cvt_hilo(v0.z, v1.z, hh, ll); KVhi[d4 * 4 + 2][kp] = hh; KVlo[d4 * 4 + 2][kp] = ll;
      cvt_hilo(v0.w, v1.w, hh, ll); KVhi[d4 * 4 + 3][kp] = hh; KVlo[d4 * 4 + 3][kp] = ll;
    }
    // P chunk (16 x 64) fp32 -> pair-packed bf16 hi/lo (converted ONCE)
#pragma unroll
    for (int s = tid; s < 256; s += 128) {
      const int r  = s >> 4;
      const int k4 = s & 15;
      const float4 v = *(const float4*)(&P[r][c * 64 + k4 * 4]);
      u32 h0, l0, h1, l1;
      cvt_hilo(v.x, v.y, h0, l0);
      cvt_hilo(v.z, v.w, h1, l1);
      const u32x2 ph = {h0, h1}, pl = {l0, l1};
      *(u32x2*)(&Phs[r][k4 * 2]) = ph;
      *(u32x2*)(&Pls[r][k4 * 2]) = pl;
    }
    __syncthreads();
#pragma unroll
    for (int slab = 0; slab < 2; ++slab) {
      const bf16x16 ahi = load_a_frag_p(&Phs[0][0] + slab * 16, 36, lane);
      const bf16x16 alo = load_a_frag_p(&Pls[0][0] + slab * 16, 36, lane);
      const bf16x16 bhi = load_b_frag_p(&KVhi[wave * 16][0] + slab * 16, 36, lane);
      const bf16x16 blo = load_b_frag_p(&KVlo[wave * 16][0] + slab * 16, 36, lane);
      wmma3(cacc, ahi, alo, bhi, blo);
    }
    __syncthreads();
  }

#pragma unroll
  for (int r = 0; r < 8; ++r) {
    const size_t row = (size_t)b * SEQ + q0 + sel * 8 + r;
    ctx[row * DMODEL + h * DK + wave * 16 + l15] = cacc[r];
  }
}

// --------------------------- layernorm --------------------------------------

__global__ __launch_bounds__(128)
void ln_kernel(const float* __restrict__ x, const float* __restrict__ g,
               const float* __restrict__ bta, float* __restrict__ out, int cols) {
  const int row  = blockIdx.x * 4 + (threadIdx.x >> 5);
  const int lane = threadIdx.x & 31;
  const float* xr = x + (size_t)row * cols;

  float s = 0.f;
  for (int i = lane; i < cols; i += 32) s += xr[i];
  const float mu = wave_sum(s) / (float)cols;

  float v = 0.f;
  for (int i = lane; i < cols; i += 32) {
    const float d = xr[i] - mu;
    v += d * d;
  }
  const float var = wave_sum(v) / (float)cols;
  const float rs  = rsqrtf(var + 1e-6f);

  float* orow = out + (size_t)row * cols;
  for (int i = lane; i < cols; i += 32)
    orow[i] = (xr[i] - mu) * rs * g[i] + bta[i];
}

// --------------------------- launcher ---------------------------------------

extern "C" void kernel_launch(void* const* d_in, const int* in_sizes, int n_in,
                              void* d_out, int out_size, void* d_ws, size_t ws_size,
                              hipStream_t stream) {
  (void)in_sizes; (void)n_in; (void)out_size; (void)ws_size;

  const float* x     = (const float*)d_in[0];
  const int*   mask  = (const int*)d_in[2];
  const float* Wq    = (const float*)d_in[3];
  const float* Wk    = (const float*)d_in[4];
  const float* Wv    = (const float*)d_in[5];
  const float* Wo    = (const float*)d_in[6];
  const float* bo    = (const float*)d_in[7];
  const float* ln1_g = (const float*)d_in[10];
  const float* ln1_b = (const float*)d_in[11];
  const float* ln2_g = (const float*)d_in[12];
  const float* ln2_b = (const float*)d_in[13];
  const float* W1    = (const float*)d_in[14];
  const float* b1    = (const float*)d_in[15];
  const float* W2    = (const float*)d_in[16];
  const float* b2    = (const float*)d_in[17];

  float* ws = (float*)d_ws;
  const size_t MD = (size_t)ROWS * DMODEL;   // 4096*512
  float* Qb = ws;
  float* Kb = Qb + MD;
  float* Vb = Kb + MD;
  float* Cb = Vb + MD;                        // ctx
  float* Hb = Cb + MD;                        // x + attn_out
  float* X1 = Hb + MD;                        // LN1 output
  float* A1 = X1 + MD;                        // gelu(x1 W1 + b1), 4096*2048
  float* H2 = A1 + (size_t)ROWS * DFF;        // x1 + ff

  float* out_x2   = (float*)d_out;            // [2,2048,512]
  float* out_attn = out_x2 + MD;              // [2,8,2048,2048]

  const dim3 blk256(256), blk128(128);
  const dim3 g512(DMODEL / 64, ROWS / 128);   // (8, 32)
  const dim3 gff (DFF / 64,   ROWS / 128);    // (32, 32)

  // Q/K/V projections
  gemm_kernel<0, false, false><<<g512, blk256, 0, stream>>>(x, Wq, nullptr, nullptr, Qb, ROWS, DMODEL, DMODEL);
  gemm_kernel<0, false, false><<<g512, blk256, 0, stream>>>(x, Wk, nullptr, nullptr, Kb, ROWS, DMODEL, DMODEL);
  gemm_kernel<0, false, false><<<g512, blk256, 0, stream>>>(x, Wv, nullptr, nullptr, Vb, ROWS, DMODEL, DMODEL);

  // attention: writes attn (268 MB) exactly once, produces ctx
  attn_kernel<<<dim3(2 * NHEADS, SEQ / 16), blk128, 0, stream>>>(Qb, Kb, Vb, mask, out_attn, Cb);

  // h = x + ctx*Wo + bo ; x1 = LN1(h)
  gemm_kernel<0, true, true><<<g512, blk256, 0, stream>>>(Cb, Wo, bo, x, Hb, ROWS, DMODEL, DMODEL);
  ln_kernel<<<ROWS / 4, blk128, 0, stream>>>(Hb, ln1_g, ln1_b, X1, DMODEL);

  // FFN with fused GELU and residual ; x2 = LN2(x1 + ff)
  gemm_kernel<1, true, false><<<gff, blk256, 0, stream>>>(X1, W1, b1, nullptr, A1, ROWS, DFF, DMODEL);
  gemm_kernel<0, true, true><<<g512, blk256, 0, stream>>>(A1, W2, b2, X1, H2, ROWS, DMODEL, DFF);
  ln_kernel<<<ROWS / 4, blk128, 0, stream>>>(H2, ln2_g, ln2_b, out_x2, DMODEL);
}
